// MultiHeadSliceSelfAttention_40776419508308
// MI455X (gfx1250) — matrix operand order repaired and checked by reading, not compiled, not run on hardware
//
#include <hip/hip_runtime.h>

typedef __bf16 v16bf __attribute__((ext_vector_type(16)));
typedef float  v8f   __attribute__((ext_vector_type(8)));
typedef unsigned int v4u __attribute__((ext_vector_type(4)));

union FragAB { v4u u[2]; v16bf v; };

#define DIM    512
#define NHEADS 8
#define HD     64
#define NTOK   1536
#define BATCH  4
#define NROWS  (BATCH * NTOK)   // 6144
#define NKT    (NTOK / 32)      // 48 key tiles of 32

__device__ __forceinline__ unsigned short f2bf(float x) {
  unsigned int u = __float_as_uint(x);
  u += 0x7fffu + ((u >> 16) & 1u);      // round-to-nearest-even
  return (unsigned short)(u >> 16);
}

__device__ __forceinline__ v8f wmma_bf16(const FragAB& a, const FragAB& b, v8f c) {
  return __builtin_amdgcn_wmma_f32_16x16x32_bf16(false, a.v, false, b.v,
                                                 (short)0, c, false, false);
}

__device__ __forceinline__ void async_copy16(unsigned lds_addr, const void* gptr) {
  unsigned long long ga = (unsigned long long)gptr;
  asm volatile("global_load_async_to_lds_b128 %0, %1, off"
               :: "v"(lds_addr), "v"(ga) : "memory");
}

// ---------------------------------------------------------------------------
// 1) concat(visual, text) + LayerNorm -> bf16 Z [6144, 512]
// ---------------------------------------------------------------------------
__global__ void ln_concat_kernel(const float* __restrict__ vis,
                                 const float* __restrict__ txt,
                                 const float* __restrict__ gamma,
                                 const float* __restrict__ beta,
                                 unsigned short* __restrict__ zln) {
  int row = blockIdx.x;                 // 0..6143
  int b = row / NTOK;
  int t = row % NTOK;
  const float* src = (t < 1024) ? (vis + ((size_t)b * 1024 + t) * DIM)
                                : (txt + ((size_t)b * 512 + (t - 1024)) * DIM);
  int tid = threadIdx.x;                // 128 threads, 4 elems each
  float4 x = *reinterpret_cast<const float4*>(src + tid * 4);
  float s = x.x + x.y + x.z + x.w;
  float q = x.x * x.x + x.y * x.y + x.z * x.z + x.w * x.w;
  #pragma unroll
  for (int m = 1; m < 32; m <<= 1) {
    s += __shfl_xor(s, m, 32);
    q += __shfl_xor(q, m, 32);
  }
  __shared__ float ss[4], sq[4];
  int w = tid >> 5, l = tid & 31;
  if (l == 0) { ss[w] = s; sq[w] = q; }
  __syncthreads();
  s = ss[0] + ss[1] + ss[2] + ss[3];
  q = sq[0] + sq[1] + sq[2] + sq[3];
  float mean = s * (1.0f / DIM);
  float var  = q * (1.0f / DIM) - mean * mean;
  float rstd = rsqrtf(var + 1e-5f);
  float4 gv = *reinterpret_cast<const float4*>(gamma + tid * 4);
  float4 bv = *reinterpret_cast<const float4*>(beta + tid * 4);
  unsigned int lo = (unsigned int)f2bf((x.x - mean) * rstd * gv.x + bv.x)
                  | ((unsigned int)f2bf((x.y - mean) * rstd * gv.y + bv.y) << 16);
  unsigned int hi = (unsigned int)f2bf((x.z - mean) * rstd * gv.z + bv.z)
                  | ((unsigned int)f2bf((x.w - mean) * rstd * gv.w + bv.w) << 16);
  unsigned long long pk = (unsigned long long)lo | ((unsigned long long)hi << 32);
  *reinterpret_cast<unsigned long long*>(zln + (size_t)row * DIM + tid * 4) = pk;
}

// ---------------------------------------------------------------------------
// 2) fp32 W [in,out] -> bf16 Wt [out,in] for Q,K,V,O (B-operand, K-contiguous)
// ---------------------------------------------------------------------------
__global__ void wt_convert_kernel(const float* __restrict__ wq,
                                  const float* __restrict__ wk,
                                  const float* __restrict__ wv,
                                  const float* __restrict__ wo,
                                  unsigned short* __restrict__ dst) {
  int wi = blockIdx.y;
  const float* W = (wi == 0) ? wq : (wi == 1) ? wk : (wi == 2) ? wv : wo;
  int idx = blockIdx.x * 256 + threadIdx.x;    // 0..262143
  int n = idx >> 9, k = idx & 511;
  dst[(size_t)wi * DIM * DIM + idx] = f2bf(W[(size_t)k * DIM + n]);
}

// ---------------------------------------------------------------------------
// 3) QKV projection GEMM: Z[6144,512] @ Wt^T + b, WMMA bf16.
//    z-dim selects Q/K/V; Q scaled by 1/8; V stored transposed [bh][d][tok].
// ---------------------------------------------------------------------------
__global__ void qkv_gemm_kernel(const unsigned short* __restrict__ zln,
                                const unsigned short* __restrict__ wt,
                                const float* __restrict__ bq,
                                const float* __restrict__ bk,
                                const float* __restrict__ bvv,
                                unsigned short* __restrict__ qb,
                                unsigned short* __restrict__ kbuf,
                                unsigned short* __restrict__ vt) {
  int which = blockIdx.z;
  const unsigned short* W = wt + (size_t)which * DIM * DIM;
  const float* bias = (which == 0) ? bq : (which == 1) ? bk : bvv;
  int tid = threadIdx.x;
  int w = tid >> 5, lane = tid & 31;
  int lmod = lane & 15, half = lane >> 4;
  int m0 = blockIdx.x * 64 + (w & 3) * 16;
  int n0 = blockIdx.y * 64 + (w >> 2) * 32;
  const unsigned short* Arow = zln + (size_t)(m0 + lmod) * DIM;
  const unsigned short* B0 = W + (size_t)(n0 + lmod) * DIM;
  const unsigned short* B1 = B0 + (size_t)16 * DIM;
  v8f c0 = {}; v8f c1 = {};
  for (int kk = 0; kk < DIM; kk += 32) {
    FragAB a, f0, f1;
    a.u[0]  = *reinterpret_cast<const v4u*>(Arow + kk + half * 16);
    a.u[1]  = *reinterpret_cast<const v4u*>(Arow + kk + half * 16 + 8);
    f0.u[0] = *reinterpret_cast<const v4u*>(B0 + kk + half * 16);
    f0.u[1] = *reinterpret_cast<const v4u*>(B0 + kk + half * 16 + 8);
    f1.u[0] = *reinterpret_cast<const v4u*>(B1 + kk + half * 16);
    f1.u[1] = *reinterpret_cast<const v4u*>(B1 + kk + half * 16 + 8);
    c0 = wmma_bf16(a, f0, c0);
    c1 = wmma_bf16(a, f1, c1);
  }
  int col0 = n0 + lmod, col1 = col0 + 16;
  float bb0 = bias[col0], bb1 = bias[col1];
  #pragma unroll
  for (int r = 0; r < 8; ++r) {
    int m = m0 + half * 8 + r;
    int bi = m / NTOK, tok = m % NTOK;
    float v0 = c0[r] + bb0;
    float v1 = c1[r] + bb1;
    int h0 = col0 >> 6, d0 = col0 & 63;
    int h1 = col1 >> 6, d1 = col1 & 63;
    if (which == 0) {            // Q, pre-scaled by 1/sqrt(hd)=1/8
      qb[((size_t)(bi * NHEADS + h0) * NTOK + tok) * HD + d0] = f2bf(v0 * 0.125f);
      qb[((size_t)(bi * NHEADS + h1) * NTOK + tok) * HD + d1] = f2bf(v1 * 0.125f);
    } else if (which == 1) {     // K [bh][tok][d]
      kbuf[((size_t)(bi * NHEADS + h0) * NTOK + tok) * HD + d0] = f2bf(v0);
      kbuf[((size_t)(bi * NHEADS + h1) * NTOK + tok) * HD + d1] = f2bf(v1);
    } else {                     // V transposed [bh][d][tok]
      vt[((size_t)(bi * NHEADS + h0) * HD + d0) * NTOK + tok] = f2bf(v0);
      vt[((size_t)(bi * NHEADS + h1) * HD + d1) * NTOK + tok] = f2bf(v1);
    }
  }
}

// ---------------------------------------------------------------------------
// 4) Flash attention with async double-buffered K/V staging.
//    All 8 waves of a WG share one (b,h): K/V tiles are staged ONCE per WG
//    into LDS via global_load_async_to_lds_b128 (2 async instrs per wave per
//    tile, ASYNCcnt retires in order -> s_wait_asynccnt 2 = previous tile
//    complete). S via 4 WMMAs, bias + online softmax, P through LDS
//    (C->A relayout), O accumulated via 4 WMMAs with per-row rescale.
// ---------------------------------------------------------------------------
__global__ void attn_kernel(const unsigned short* __restrict__ qb,
                            const unsigned short* __restrict__ kbuf,
                            const unsigned short* __restrict__ vt,
                            const float* __restrict__ btab,
                            unsigned short* __restrict__ attn) {
  __shared__ unsigned short klds[2][32 * HD] __attribute__((aligned(16)));  // [buf][key][d]
  __shared__ unsigned short vlds[2][HD * 32] __attribute__((aligned(16)));  // [buf][d][key]
  __shared__ unsigned short plds[8][16][32] __attribute__((aligned(16)));
  int bh = blockIdx.x;
  int b = bh >> 3, h = bh & 7;
  int tid = threadIdx.x, w = tid >> 5, lane = tid & 31;
  int lmod = lane & 15, half = lane >> 4;
  int qbase = blockIdx.y * 128 + w * 16;
  const unsigned short* Qp = qb   + (size_t)bh * NTOK * HD;
  const unsigned short* Kp = kbuf + (size_t)bh * NTOK * HD;
  const unsigned short* Vp = vt   + (size_t)bh * HD * NTOK;

  // cooperative staging roles: each thread moves 16B of K and 16B of V per tile
  const unsigned short* kg = Kp + tid * 8;                    // + kb0*HD per tile
  int vd = tid >> 2, vc = tid & 3;
  const unsigned short* vg = Vp + (size_t)vd * NTOK + vc * 8; // + kb0 per tile
  unsigned klA[2], vlA[2];
  klA[0] = (unsigned)(size_t)&klds[0][tid * 8];
  klA[1] = (unsigned)(size_t)&klds[1][tid * 8];
  vlA[0] = (unsigned)(size_t)&vlds[0][vd * 32 + vc * 8];
  vlA[1] = (unsigned)(size_t)&vlds[1][vd * 32 + vc * 8];

  // prefetch tile 0 into buffer 0
  async_copy16(klA[0], kg);
  async_copy16(vlA[0], vg);

  const unsigned short* Qrow = Qp + (size_t)(qbase + lmod) * HD;
  FragAB aq0, aq1;                         // Q 16x64, dims split 0..31 / 32..63
  aq0.u[0] = *reinterpret_cast<const v4u*>(Qrow + half * 16);
  aq0.u[1] = *reinterpret_cast<const v4u*>(Qrow + half * 16 + 8);
  aq1.u[0] = *reinterpret_cast<const v4u*>(Qrow + 32 + half * 16);
  aq1.u[1] = *reinterpret_cast<const v4u*>(Qrow + 32 + half * 16 + 8);

  v8f o0 = {}, o1 = {}, o2 = {}, o3 = {};
  float mrun[8], lrun[8];
  #pragma unroll
  for (int r = 0; r < 8; ++r) { mrun[r] = -1e30f; lrun[r] = 0.0f; }

  for (int j = 0; j < NKT; ++j) {
    int cur = j & 1;
    int kb0 = j * 32;
    if (j + 1 < NKT) {                     // prefetch next tile into spare buffer
      int nxt = cur ^ 1;
      async_copy16(klA[nxt], kg + (size_t)(kb0 + 32) * HD);
      async_copy16(vlA[nxt], vg + (kb0 + 32));
      asm volatile("s_wait_asynccnt 0x2" ::: "memory");  // tile j's K+V landed
    } else {
      asm volatile("s_wait_asynccnt 0x0" ::: "memory");
    }
    __syncthreads();                       // staging visible to all waves

    // ---- S = Q @ K^T for 32 keys (two 16-key column sets), K from LDS ----
    const unsigned short* Kc0 = &klds[cur][lmod * HD];
    const unsigned short* Kc1 = &klds[cur][(16 + lmod) * HD];
    FragAB k00, k01, k10, k11;
    k00.u[0] = *reinterpret_cast<const v4u*>(Kc0 + half * 16);
    k00.u[1] = *reinterpret_cast<const v4u*>(Kc0 + half * 16 + 8);
    k01.u[0] = *reinterpret_cast<const v4u*>(Kc0 + 32 + half * 16);
    k01.u[1] = *reinterpret_cast<const v4u*>(Kc0 + 32 + half * 16 + 8);
    k10.u[0] = *reinterpret_cast<const v4u*>(Kc1 + half * 16);
    k10.u[1] = *reinterpret_cast<const v4u*>(Kc1 + half * 16 + 8);
    k11.u[0] = *reinterpret_cast<const v4u*>(Kc1 + 32 + half * 16);
    k11.u[1] = *reinterpret_cast<const v4u*>(Kc1 + 32 + half * 16 + 8);
    v8f s0 = {}, s1 = {};
    s0 = wmma_bf16(aq0, k00, s0);
    s0 = wmma_bf16(aq1, k01, s0);
    s1 = wmma_bf16(aq0, k10, s1);
    s1 = wmma_bf16(aq1, k11, s1);

    // ---- bias + online softmax (row = half*8+r across 16 lanes) ----
    float p0[8], p1[8], alpha[8];
    #pragma unroll
    for (int r = 0; r < 8; ++r) {
      int qrow = qbase + half * 8 + r;
      int rel0 = qrow - (kb0 + lmod) + 1023;
      int rel1 = rel0 - 16;
      rel0 = min(max(rel0, 0), 2046);
      rel1 = min(max(rel1, 0), 2046);
      float t0 = s0[r] + btab[rel0 * NHEADS + h];
      float t1 = s1[r] + btab[rel1 * NHEADS + h];
      float mx = fmaxf(t0, t1);
      #pragma unroll
      for (int m = 1; m < 16; m <<= 1) mx = fmaxf(mx, __shfl_xor(mx, m, 16));
      float mnew = fmaxf(mrun[r], mx);
      float al = __expf(mrun[r] - mnew);
      float e0 = __expf(t0 - mnew);
      float e1 = __expf(t1 - mnew);
      float rs = e0 + e1;
      #pragma unroll
      for (int m = 1; m < 16; m <<= 1) rs += __shfl_xor(rs, m, 16);
      lrun[r] = lrun[r] * al + rs;
      mrun[r] = mnew;
      alpha[r] = al;
      p0[r] = e0;
      p1[r] = e1;
    }

    // ---- P: C-layout -> A-layout via per-wave LDS slab ----
    #pragma unroll
    for (int r = 0; r < 8; ++r) {
      plds[w][half * 8 + r][lmod]      = f2bf(p0[r]);
      plds[w][half * 8 + r][16 + lmod] = f2bf(p1[r]);
    }
    asm volatile("" ::: "memory");   // keep store->load order; LDS is in-order per wave
    FragAB ap;
    ap.u[0] = *reinterpret_cast<const v4u*>(&plds[w][lmod][half * 16]);
    ap.u[1] = *reinterpret_cast<const v4u*>(&plds[w][lmod][half * 16 + 8]);

    // ---- O = O*alpha + P @ V (4 d-slices of 16), V from LDS ----
    #pragma unroll
    for (int r = 0; r < 8; ++r) {
      o0[r] *= alpha[r]; o1[r] *= alpha[r]; o2[r] *= alpha[r]; o3[r] *= alpha[r];
    }
    FragAB bv;
    const unsigned short* Vc = &vlds[cur][lmod * 32];
    bv.u[0] = *reinterpret_cast<const v4u*>(Vc + half * 16);
    bv.u[1] = *reinterpret_cast<const v4u*>(Vc + half * 16 + 8);
    o0 = wmma_bf16(ap, bv, o0);
    Vc += 16 * 32;
    bv.u[0] = *reinterpret_cast<const v4u*>(Vc + half * 16);
    bv.u[1] = *reinterpret_cast<const v4u*>(Vc + half * 16 + 8);
    o1 = wmma_bf16(ap, bv, o1);
    Vc += 16 * 32;
    bv.u[0] = *reinterpret_cast<const v4u*>(Vc + half * 16);
    bv.u[1] = *reinterpret_cast<const v4u*>(Vc + half * 16 + 8);
    o2 = wmma_bf16(ap, bv, o2);
    Vc += 16 * 32;
    bv.u[0] = *reinterpret_cast<const v4u*>(Vc + half * 16);
    bv.u[1] = *reinterpret_cast<const v4u*>(Vc + half * 16 + 8);
    o3 = wmma_bf16(ap, bv, o3);

    __syncthreads();                 // all waves done with buffer 'cur'
  }

  // ---- epilogue: normalize and store bf16 attn output [B,N,D] ----
  #pragma unroll
  for (int r = 0; r < 8; ++r) lrun[r] = 1.0f / lrun[r];
  unsigned short* dst = attn + ((size_t)b * NTOK + qbase) * DIM + h * HD;
  #pragma unroll
  for (int r = 0; r < 8; ++r) {
    unsigned short* d = dst + (size_t)(half * 8 + r) * DIM;
    d[lmod]      = f2bf(o0[r] * lrun[r]);
    d[16 + lmod] = f2bf(o1[r] * lrun[r]);
    d[32 + lmod] = f2bf(o2[r] * lrun[r]);
    d[48 + lmod] = f2bf(o3[r] * lrun[r]);
  }
}

// ---------------------------------------------------------------------------
// 5) Output projection: attn[6144,512] @ Wt_O^T + b_O -> fp32 out
// ---------------------------------------------------------------------------
__global__ void out_gemm_kernel(const unsigned short* __restrict__ attn,
                                const unsigned short* __restrict__ wto,
                                const float* __restrict__ bo,
                                float* __restrict__ out) {
  int tid = threadIdx.x;
  int w = tid >> 5, lane = tid & 31;
  int lmod = lane & 15, half = lane >> 4;
  int m0 = blockIdx.x * 64 + (w & 3) * 16;
  int n0 = blockIdx.y * 64 + (w >> 2) * 32;
  const unsigned short* Arow = attn + (size_t)(m0 + lmod) * DIM;
  const unsigned short* B0 = wto + (size_t)(n0 + lmod) * DIM;
  const unsigned short* B1 = B0 + (size_t)16 * DIM;
  v8f c0 = {}; v8f c1 = {};
  for (int kk = 0; kk < DIM; kk += 32) {
    FragAB a, f0, f1;
    a.u[0]  = *reinterpret_cast<const v4u*>(Arow + kk + half * 16);
    a.u[1]  = *reinterpret_cast<const v4u*>(Arow + kk + half * 16 + 8);
    f0.u[0] = *reinterpret_cast<const v4u*>(B0 + kk + half * 16);
    f0.u[1] = *reinterpret_cast<const v4u*>(B0 + kk + half * 16 + 8);
    f1.u[0] = *reinterpret_cast<const v4u*>(B1 + kk + half * 16);
    f1.u[1] = *reinterpret_cast<const v4u*>(B1 + kk + half * 16 + 8);
    c0 = wmma_bf16(a, f0, c0);
    c1 = wmma_bf16(a, f1, c1);
  }
  int col0 = n0 + lmod, col1 = col0 + 16;
  float bb0 = bo[col0], bb1 = bo[col1];
  #pragma unroll
  for (int r = 0; r < 8; ++r) {
    int m = m0 + half * 8 + r;
    out[(size_t)m * DIM + col0] = c0[r] + bb0;
    out[(size_t)m * DIM + col1] = c1[r] + bb1;
  }
}

// ---------------------------------------------------------------------------
extern "C" void kernel_launch(void* const* d_in, const int* in_sizes, int n_in,
                              void* d_out, int out_size, void* d_ws, size_t ws_size,
                              hipStream_t stream) {
  const float* vis   = (const float*)d_in[0];
  const float* txt   = (const float*)d_in[1];
  const float* WQ    = (const float*)d_in[2];
  const float* bQ    = (const float*)d_in[3];
  const float* WK    = (const float*)d_in[4];
  const float* bK    = (const float*)d_in[5];
  const float* WV    = (const float*)d_in[6];
  const float* bV    = (const float*)d_in[7];
  const float* WO    = (const float*)d_in[8];
  const float* bO    = (const float*)d_in[9];
  const float* btab  = (const float*)d_in[10];
  const float* gamma = (const float*)d_in[11];
  const float* beta  = (const float*)d_in[12];
  float* out = (float*)d_out;

  unsigned short* ws  = (unsigned short*)d_ws;
  unsigned short* zln = ws;                               // 6144*512
  unsigned short* wt  = zln + (size_t)NROWS * DIM;        // 4*512*512 (Q,K,V,O)
  unsigned short* qb  = wt  + (size_t)4 * DIM * DIM;      // [bh][tok][d]
  unsigned short* kb  = qb  + (size_t)NROWS * DIM;        // [bh][tok][d]
  unsigned short* vt  = kb  + (size_t)NROWS * DIM;        // [bh][d][tok]
  unsigned short* att = vt  + (size_t)NROWS * DIM;        // [B,N,D]

  ln_concat_kernel<<<dim3(NROWS), dim3(128), 0, stream>>>(vis, txt, gamma, beta, zln);
  wt_convert_kernel<<<dim3(1024, 4), dim3(256), 0, stream>>>(WQ, WK, WV, WO, wt);
  qkv_gemm_kernel<<<dim3(NROWS / 64, DIM / 64, 3), dim3(256), 0, stream>>>(
      zln, wt, bQ, bK, bV, qb, kb, vt);
  attn_kernel<<<dim3(BATCH * NHEADS, NTOK / 128), dim3(256), 0, stream>>>(
      qb, kb, vt, btab, att);
  out_gemm_kernel<<<dim3(NROWS / 64, DIM / 64), dim3(256), 0, stream>>>(
      att, wt + (size_t)3 * DIM * DIM, bO, out);
}
